// Patcher_17480516895526
// MI455X (gfx1250) — compile-verified, matrix operand order checked by reading, not run on hardware
//
#include <hip/hip_runtime.h>
#include <hip/hip_bf16.h>
#include <math.h>

#define Pp   128
#define PPc  (Pp * Pp)
#define WINc 256
#define Hc   896
#define Wc   896
#define Bc   8
#define Nc   24
#define HPAD (Hc + WINc)
#define WPAD (Wc + WINc)

typedef __attribute__((ext_vector_type(16))) _Float16 v16h;
typedef __attribute__((ext_vector_type(8)))  float    v8f;

// sigmoid via v_exp_f32 + v_rcp_f32 (2 trans ops, no IEEE div sequence)
__device__ __forceinline__ float sigmoid_fast(float x) {
    return __builtin_amdgcn_rcpf(1.0f + __expf(-x));
}

// ---------------------------------------------------------------------------
// Stage 0: per-box parameters (scale regressor + Patcher.create) + zero masksum
// pbox layout per box, stride 8 floats: [ymp, xmp, ps, scale, valid, f0, f1, 0]
// ---------------------------------------------------------------------------
__global__ void k_params(const float* __restrict__ boxes, const float* __restrict__ Ws,
                         const float* __restrict__ bs, float* __restrict__ pbox,
                         float* __restrict__ masksum) {
    int t = blockIdx.x * blockDim.x + threadIdx.x;
    if (t >= Bc * Nc) return;
    masksum[t] = 0.0f;                       // ws is poisoned 0xAA; init accumulators
    const float* bx = boxes + t * 4;
    float y0 = bx[0], x0 = bx[1], y1 = bx[2], x1 = bx[3];
    float h = y1 - y0, w = x1 - x0;
    float f0 = h / (float)Hc, f1 = w / (float)Wc;
    float scale = sigmoid_fast(f0 * Ws[0] + f1 * Ws[1] + bs[0]) * 0.4f;
    float ps = floorf(sqrtf(h * w * scale));
    float oy = y0 + h * 0.5f, ox = x0 + w * 0.5f;
    float ymp = fmaxf(oy - ps * 0.5f, 0.0f);
    float xmp = fmaxf(ox - ps * 0.5f, 0.0f);
    if (ymp + ps > (float)Hc) ymp = (float)Hc - ps;
    if (xmp + ps > (float)Wc) xmp = (float)Wc - ps;
    float* o = pbox + t * 8;
    o[0] = ymp; o[1] = xmp; o[2] = ps; o[3] = scale;
    o[4] = (ps > 60.0f) ? 1.0f : 0.0f;       // MINH
    o[5] = f0;  o[6] = f1;  o[7] = 0.0f;
}

// ---------------------------------------------------------------------------
// Stage 1: patch channel stats (mu_p, sd_p = std + 1e-6) via LDS atomics
// ---------------------------------------------------------------------------
__global__ void k_pstats(const float* __restrict__ patch, float* __restrict__ pstats) {
    __shared__ float acc[6];
    if (threadIdx.x < 6) acc[threadIdx.x] = 0.0f;
    __syncthreads();
    float s[3] = {0, 0, 0}, q[3] = {0, 0, 0};
    for (int i = threadIdx.x; i < PPc; i += blockDim.x)
        for (int ch = 0; ch < 3; ++ch) {
            float v = patch[i * 3 + ch];
            s[ch] += v; q[ch] += v * v;
        }
    for (int ch = 0; ch < 3; ++ch) {
        atomicAdd(&acc[ch], s[ch]);
        atomicAdd(&acc[3 + ch], q[ch]);
    }
    __syncthreads();
    if (threadIdx.x < 3) {
        int ch = threadIdx.x;
        float mu  = acc[ch] / (float)PPc;
        float var = fmaxf(acc[3 + ch] / (float)PPc - mu * mu, 0.0f);
        pstats[ch]     = mu;
        pstats[3 + ch] = sqrtf(var) + 1e-6f;
    }
}

// ---------------------------------------------------------------------------
// Stage 2: mask generator matmul via WMMA.
// masksum[b,n] = sum_p sigmoid(feats[b,n,:] . Wg[:,p] + bg[p])
// Tile: M=16 boxes, N=16 pixels, K=32 (only k=0,1 nonzero).
// Grid: 16 blocks = (b, mtile); 8 waves/block each cover 128 of 1024 ntiles.
// ---------------------------------------------------------------------------
__global__ void k_masksum(const float* __restrict__ pbox, const float* __restrict__ Wg,
                          const float* __restrict__ bg, float* __restrict__ masksum) {
    int b     = blockIdx.x >> 1;
    int mtile = blockIdx.x & 1;
    int lane  = threadIdx.x & 31;
    int wave  = threadIdx.x >> 5;

    // A fragment: 16x32 f16. Lanes 0-15 hold K=0..7 in halves 0..7 (we use K=0,1).
    v16h a = {};
    if (lane < 16) {
        int box = mtile * 16 + lane;
        if (box < Nc) {
            const float* pb = pbox + (b * Nc + box) * 8;
            a[0] = (_Float16)pb[5];
            a[1] = (_Float16)pb[6];
        }
    }

    float acc[8] = {0, 0, 0, 0, 0, 0, 0, 0};
#pragma unroll 2
    for (int nt = wave; nt < PPc / 16; nt += 8) {
        int p = nt * 16 + (lane & 15);
        // B fragment: 32x16 f16. Lanes 0-15 hold K=0..7 in halves 0..7 (we use K=0,1).
        v16h bm = {};
        if (lane < 16) {
            bm[0] = (_Float16)Wg[p];
            bm[1] = (_Float16)Wg[PPc + p];
        }
        float bias = bg[p];                  // bias depends only on column
        v8f c;
        for (int v = 0; v < 8; ++v) c[v] = bias;
        c = __builtin_amdgcn_wmma_f32_16x16x32_f16(
                /*neg_a=*/false, a, /*neg_b=*/false, bm,
                /*c_mod=*/(short)0, c, /*reuse_a=*/false, /*reuse_b=*/false);
        for (int v = 0; v < 8; ++v) acc[v] += sigmoid_fast(c[v]);
    }
    // Row sums: reduce over the 16 lanes holding different columns of the same rows.
    for (int off = 1; off < 16; off <<= 1)
        for (int v = 0; v < 8; ++v) acc[v] += __shfl_xor(acc[v], off, 32);
    if ((lane & 15) == 0) {
        int rbase = mtile * 16 + ((lane >= 16) ? 8 : 0);
        for (int v = 0; v < 8; ++v) {
            int box = rbase + v;
            if (box < Nc) atomicAdd(&masksum[b * Nc + box], acc[v]);
        }
    }
}

// ---------------------------------------------------------------------------
// Stage 3: the sequential per-image scan. One workgroup (1024 thr = 32 waves)
// per image, one image per WGP. The P*P*3 f32 background crop (192 KB) and a
// 2 KB mask bitmask live in LDS (CDNA5: 320 KB/WGP) -- all phase-B gathers are
// LDS hits; only the final image write and 12 patch loads touch global memory.
// ---------------------------------------------------------------------------
__global__ void __launch_bounds__(1024)
k_scan(float* __restrict__ out, const float* __restrict__ pbox,
       const float* __restrict__ patch, const float* __restrict__ pstats,
       const float* __restrict__ Wg, const float* __restrict__ bg) {
    __shared__ float        spbg[PPc * 3];      // 196608 B: background crop
    __shared__ unsigned int smask[PPc / 32];    //   2048 B: mask>0.5 bitmask
    __shared__ float        sacc[6];
    __shared__ float        smu[3], ssd[3];

    int b = blockIdx.x;
    float* img = out + (size_t)b * Hc * Wc * 3;
    int lane = threadIdx.x & 31;
    int wave = threadIdx.x >> 5;

    float mp[3], rsp[3];
    for (int ch = 0; ch < 3; ++ch) { mp[ch] = pstats[ch]; rsp[ch] = 1.0f / pstats[3 + ch]; }

    for (int n = 0; n < Nc; ++n) {
        const float* pb = pbox + (b * Nc + n) * 8;
        float ymp = pb[0], xmp = pb[1], ps = pb[2];
        float f0 = pb[5], f1 = pb[6];
        if (pb[4] == 0.0f) continue;         // invalid box: no image write, no bgl
        int yi = (int)ymp, xi = (int)xmp;
        int ph = (int)ps, pw = ph;
        float phf = (float)(ph > 1 ? ph : 1), pwf = phf;
        float fwd = phf * (1.0f / (float)Pp);   // crop step    (phase A)
        float bwd = (float)Pp / phf;            // resize step  (phase B), uniform div

        // Warm the window into cache while we compute (global_prefetch_b8).
        if ((int)threadIdx.x < ph)
            __builtin_prefetch(img + ((size_t)(yi + (int)threadIdx.x) * Wc + xi) * 3, 0, 1);

        if (threadIdx.x < 6) sacc[threadIdx.x] = 0.0f;
        __syncthreads();

        // ---- Phase A: bilinear crop -> LDS, stats, and mask bitmask ----
        float s0 = 0, s1 = 0, s2 = 0, q0 = 0, q1 = 0, q2 = 0;
        for (int py = wave; py < Pp; py += 32) {
            float ysf = (float)yi + ((float)py + 0.5f) * fwd - 0.5f;
            ysf = fminf(fmaxf(ysf, 0.0f), (float)(HPAD - 1));
            int y0 = (int)floorf(ysf);
            int y1 = (y0 + 1 < HPAD - 1) ? y0 + 1 : HPAD - 1;
            float wy = ysf - (float)y0;
            const float* r0 = img + ((size_t)y0 * Wc) * 3;
            const float* r1 = img + ((size_t)y1 * Wc) * 3;
#pragma unroll
            for (int i = 0; i < Pp / 32; ++i) {
                int px = lane + i * 32;
                int p  = py * Pp + px;
                float xsf = (float)xi + ((float)px + 0.5f) * fwd - 0.5f;
                xsf = fminf(fmaxf(xsf, 0.0f), (float)(WPAD - 1));
                int x0 = (int)floorf(xsf);
                int x1 = (x0 + 1 < WPAD - 1) ? x0 + 1 : WPAD - 1;
                float wx = xsf - (float)x0;
                bool i00 = (y0 < Hc) & (x0 < Wc), i01 = (y0 < Hc) & (x1 < Wc);
                bool i10 = (y1 < Hc) & (x0 < Wc), i11 = (y1 < Hc) & (x1 < Wc);
                float w00 = (1 - wy) * (1 - wx), w01 = (1 - wy) * wx;
                float w10 = wy * (1 - wx),       w11 = wy * wx;
                float vv[3];
                for (int ch = 0; ch < 3; ++ch) {
                    float v00 = i00 ? r0[x0 * 3 + ch] : 0.0f;
                    float v01 = i01 ? r0[x1 * 3 + ch] : 0.0f;
                    float v10 = i10 ? r1[x0 * 3 + ch] : 0.0f;
                    float v11 = i11 ? r1[x1 * 3 + ch] : 0.0f;
                    vv[ch] = w00 * v00 + w01 * v01 + w10 * v10 + w11 * v11;
                    spbg[p * 3 + ch] = vv[ch];
                }
                s0 += vv[0]; s1 += vv[1]; s2 += vv[2];
                q0 += vv[0] * vv[0]; q1 += vv[1] * vv[1]; q2 += vv[2] * vv[2];
                // Mask bit: sigmoid(z)>0.5 <=> z>0. One ballot per 32 pixels.
                float mlin = f0 * Wg[p] + f1 * Wg[PPc + p] + bg[p];
                unsigned long long bal = __ballot(mlin > 0.0f);
                if (lane == 0) smask[py * (Pp / 32) + i] = (unsigned int)bal;
            }
        }
        atomicAdd(&sacc[0], s0); atomicAdd(&sacc[1], s1); atomicAdd(&sacc[2], s2);
        atomicAdd(&sacc[3], q0); atomicAdd(&sacc[4], q1); atomicAdd(&sacc[5], q2);
        __syncthreads();
        if (threadIdx.x < 3) {
            int ch = threadIdx.x;
            float mu  = sacc[ch] / (float)PPc;
            float var = fmaxf(sacc[3 + ch] / (float)PPc - mu * mu, 0.0f);
            smu[ch] = mu;
            ssd[ch] = sqrtf(var) + 1e-6f;
        }
        __syncthreads();
        float mb[3], sb[3];
        for (int ch = 0; ch < 3; ++ch) { mb[ch] = smu[ch]; sb[ch] = ssd[ch]; }

        // ---- Phase B: bilinear resize of thresholded blend into the image ----
        for (int r = wave; r < ph; r += 32) {
            float syf = ((float)r + 0.5f) * bwd - 0.5f;
            syf = fminf(fmaxf(syf, 0.0f), (float)(Pp - 1));
            int sy0 = (int)floorf(syf);
            int sy1 = (sy0 + 1 < Pp - 1) ? sy0 + 1 : Pp - 1;
            float wy = syf - (float)sy0;
            for (int c = lane; c < pw; c += 32) {
                float sxf = ((float)c + 0.5f) * bwd - 0.5f;
                sxf = fminf(fmaxf(sxf, 0.0f), (float)(Pp - 1));
                int sx0 = (int)floorf(sxf);
                int sx1 = (sx0 + 1 < Pp - 1) ? sx0 + 1 : Pp - 1;
                float wx = sxf - (float)sx0;
                int ysel[2] = {sy0, sy1}; int xsel[2] = {sx0, sx1};
                float wys[2] = {1 - wy, wy}, wxs[2] = {1 - wx, wx};
                float o[3] = {0, 0, 0};
#pragma unroll
                for (int iy = 0; iy < 2; ++iy)
#pragma unroll
                    for (int ix = 0; ix < 2; ++ix) {
                        int p = ysel[iy] * Pp + xsel[ix];
                        float wgt = wys[iy] * wxs[ix];
                        bool m = (smask[p >> 5] >> (p & 31)) & 1u;
                        if (m) {
                            for (int ch = 0; ch < 3; ++ch)
                                o[ch] += wgt * ((patch[p * 3 + ch] - mp[ch]) * rsp[ch] * sb[ch] + mb[ch]);
                        } else {
                            for (int ch = 0; ch < 3; ++ch)
                                o[ch] += wgt * spbg[p * 3 + ch];
                        }
                    }
                float* d = img + ((size_t)(yi + r) * Wc + (xi + c)) * 3;
                d[0] = o[0]; d[1] = o[1]; d[2] = o[2];
            }
        }
        __threadfence();   // box n global writes visible to box n+1 reads (same WGP)
        __syncthreads();
    }
}

// ---------------------------------------------------------------------------
// Stage 4: scalar loss = sum_b ( bgl + mean(valid scales) + 0.5*var )
// ---------------------------------------------------------------------------
__global__ void k_loss(const float* __restrict__ pbox, const float* __restrict__ masksum,
                       float* __restrict__ out_loss) {
    if (threadIdx.x != 0 || blockIdx.x != 0) return;
    float total = 0.0f;
    for (int b = 0; b < Bc; ++b) {
        float bgl = 0, ssum = 0, cnt = 0;
        for (int n = 0; n < Nc; ++n) {
            const float* pb = pbox + (b * Nc + n) * 8;
            if (pb[4] > 0.5f) {
                bgl += masksum[b * Nc + n] / (float)PPc;
                ssum += pb[3];
                cnt += 1.0f;
            }
        }
        float nn = fmaxf(cnt, 1.0f);
        float m = ssum / nn;
        float var = 0.0f;
        for (int n = 0; n < Nc; ++n) {
            const float* pb = pbox + (b * Nc + n) * 8;
            if (pb[4] > 0.5f) { float d = pb[3] - m; var += d * d; }
        }
        var /= nn;
        total += bgl + m + 0.5f * var;
    }
    *out_loss = total;
}

// ---------------------------------------------------------------------------
extern "C" void kernel_launch(void* const* d_in, const int* in_sizes, int n_in,
                              void* d_out, int out_size, void* d_ws, size_t ws_size,
                              hipStream_t stream) {
    (void)in_sizes; (void)n_in; (void)out_size; (void)ws_size;
    const float* boxes  = (const float*)d_in[0];
    const float* images = (const float*)d_in[1];
    const float* patch  = (const float*)d_in[2];
    const float* Ws     = (const float*)d_in[3];
    const float* bs     = (const float*)d_in[4];
    const float* Wg     = (const float*)d_in[5];
    const float* bg     = (const float*)d_in[6];
    float* out = (float*)d_out;
    float* ws  = (float*)d_ws;

    float* pbox    = ws;           // 192 * 8 = 1536 floats
    float* pstats  = ws + 1536;    // 6 (+2 pad)
    float* masksum = ws + 1544;    // 192 floats   (total ws use < 8 KB)

    // Work in place on d_out: seed it with the input images.
    hipMemcpyAsync(out, images, (size_t)Bc * Hc * Wc * 3 * sizeof(float),
                   hipMemcpyDeviceToDevice, stream);

    k_params <<<1, 256, 0, stream>>>(boxes, Ws, bs, pbox, masksum);
    k_pstats <<<1, 256, 0, stream>>>(patch, pstats);
    k_masksum<<<16, 256, 0, stream>>>(pbox, Wg, bg, masksum);
    k_scan   <<<Bc, 1024, 0, stream>>>(out, pbox, patch, pstats, Wg, bg);
    k_loss   <<<1, 32, 0, stream>>>(pbox, masksum, out + (size_t)Bc * Hc * Wc * 3);
}
